// Token_Embed_29265907155065
// MI455X (gfx1250) — matrix operand order repaired and checked by reading, compiled, not run on hardware
//
#include <hip/hip_runtime.h>
#include <hip/hip_bf16.h>
#include <stdint.h>

// ---------------- problem constants ----------------
#define BDIM  4
#define NPT   100000
#define BN    (BDIM * NPT)      // 400000 points
#define NSP   512
#define SEGS  (BDIM * NSP)      // 2048 global segments
#define FEATN 11
#define H2    256               // h channels (after fused conv1*conv2)
#define H3    512
#define EOUT  384

typedef __bf16 bf16_t;
typedef bf16_t v16bf __attribute__((ext_vector_type(16)));
typedef float  v8f   __attribute__((ext_vector_type(8)));
typedef unsigned int v4u __attribute__((ext_vector_type(4)));

union FragU { v16bf bf; v4u q[2]; unsigned short us[16]; };

#if __has_builtin(__builtin_amdgcn_global_load_async_to_lds_b128) && \
    __has_builtin(__builtin_amdgcn_s_wait_asynccnt)
#define USE_ASYNC_LDS 1
// builtin expects: ( __device__ v4i*, __shared__ v4i*, i32 offset, i32 cpol )
typedef int gv4i __attribute__((vector_size(16)));
typedef __attribute__((address_space(1))) gv4i* as1_v4i_ptr;
typedef __attribute__((address_space(3))) gv4i* as3_v4i_ptr;
#endif

__device__ __forceinline__ unsigned short f2bf(float f) {
  unsigned int u = __float_as_uint(f);
  u += 0x7fffu + ((u >> 16) & 1u);          // round-to-nearest-even
  return (unsigned short)(u >> 16);
}

__device__ __forceinline__ v8f wmma_bf16(v16bf a, v16bf b, v8f c) {
  // D = A(16x32) * B(32x16) + C, fp32 accumulate
  return __builtin_amdgcn_wmma_f32_16x16x32_bf16(false, a, false, b, (short)0, c,
                                                 false, false);
}

// B fragment: weights stored row-major [N][K] bf16; lane n = ncol + (lane&15),
// elems 0-7 <-> K = k0+8*hi..+7, elems 8-15 <-> K = k0+16+8*hi..+7 (2 x b128).
__device__ __forceinline__ v16bf load_bfrag(const unsigned short* W, int K,
                                            int ncol, int k0, int lane) {
  int m = lane & 15, hi = lane >> 4;
  FragU f;
  const unsigned short* p = W + (size_t)(ncol + m) * K + k0 + 8 * hi;
  f.q[0] = *(const v4u*)p;
  f.q[1] = *(const v4u*)(p + 16);
  return f.bf;
}

// A fragment: row-major [rows][K] bf16; lane row = caller's (base + lane&15),
// elems 0-7 <-> K = k0+8*hi..+7, elems 8-15 <-> K = k0+16+8*hi..+7.
__device__ __forceinline__ v16bf load_afrag(const unsigned short* A, int K,
                                            size_t row, int k0, int lane) {
  int hi = lane >> 4;
  FragU f;
  f.q[0] = *(const v4u*)(A + row * (size_t)K + k0 + 8 * hi);
  f.q[1] = *(const v4u*)(A + row * (size_t)K + k0 + 16 + 8 * hi);
  return f.bf;
}

// ---------------- tiny prep kernels ----------------
__global__ void k_zero(int* p, int n) {
  int t = blockIdx.x * blockDim.x + threadIdx.x;
  if (t < n) p[t] = 0;
}

__global__ void k_count(const int* __restrict__ sidx, int* __restrict__ counts) {
  int t = blockIdx.x * blockDim.x + threadIdx.x;
  if (t >= BN) return;
  int b = t / NPT;
  atomicAdd(&counts[sidx[t] + b * NSP], 1);
}

__global__ void k_scan(const int* __restrict__ counts, int* __restrict__ offsets) {
  __shared__ int lds[257];
  int t = threadIdx.x;               // 256 threads, 8 segs each
  int base = t * 8, s = 0;
#pragma unroll
  for (int j = 0; j < 8; ++j) s += counts[base + j];
  lds[t] = s;
  __syncthreads();
  if (t == 0) {
    int run = 0;
    for (int i = 0; i < 256; ++i) { int tmp = lds[i]; lds[i] = run; run += tmp; }
    lds[256] = run;
  }
  __syncthreads();
  int run = lds[t];
#pragma unroll
  for (int j = 0; j < 8; ++j) { offsets[base + j] = run; run += counts[base + j]; }
  if (t == 255) offsets[SEGS] = lds[256];
}

__global__ void k_scatter(const int* __restrict__ sidx, const int* __restrict__ offsets,
                          int* __restrict__ cursor, int* __restrict__ order,
                          int* __restrict__ segsorted) {
  int t = blockIdx.x * blockDim.x + threadIdx.x;
  if (t >= BN) return;
  int b = t / NPT;
  int seg = sidx[t] + b * NSP;
  int pos = offsets[seg] + atomicAdd(&cursor[seg], 1);
  order[pos] = t;
  segsorted[pos] = seg;
}

// Wc[n][k] = sum_j W2[n][j]*W1[j][k]  (K padded 11->32), bc = b2 + W2@b1
__global__ void k_fuse12(const float* __restrict__ W1, const float* __restrict__ b1,
                         const float* __restrict__ W2, const float* __restrict__ b2,
                         unsigned short* __restrict__ Wc, float* __restrict__ bc) {
  int t = blockIdx.x * blockDim.x + threadIdx.x;
  if (t >= 256 * 32) return;
  int n = t >> 5, k = t & 31;
  float s = 0.f;
  if (k < FEATN)
    for (int j = 0; j < 128; ++j) s += W2[n * 128 + j] * W1[j * FEATN + k];
  Wc[n * 32 + k] = (k < FEATN) ? f2bf(s) : (unsigned short)0;
  if (k == 0) {
    float v = b2[n];
    for (int j = 0; j < 128; ++j) v += W2[n * 128 + j] * b1[j];
    bc[n] = v;
  }
}

__global__ void k_cvt(const float* __restrict__ src, int stride, int cols, int total,
                      unsigned short* __restrict__ dst) {
  int t = blockIdx.x * blockDim.x + threadIdx.x;
  if (t >= total) return;
  int r = t / cols, c = t - r * cols;
  dst[t] = f2bf(src[(size_t)r * stride + c]);
}

// ---------------- stage 1: h = x @ Wc.T + bc, per-segment, fused segmax ----
// One WG (4 waves) per segment; waves partition the 256 output channels
// (4 n-tiles each); all waves walk all 16-row tiles of the segment's list.
// Tail rows are CLAMPED to row cnt-1: duplicate D-rows equal the genuine last
// row, so max is unaffected and address-clamped rewrites are benign -> the
// whole epilogue is branch-free (no exec-mask saveexec chains).
__global__ void __launch_bounds__(128) k_stage1(
    const float* __restrict__ feats, const int* __restrict__ order,
    const int* __restrict__ offsets, const int* __restrict__ counts,
    const unsigned short* __restrict__ Wc, const float* __restrict__ bc,
    unsigned short* __restrict__ hsorted, unsigned short* __restrict__ gmax) {
  int seg = blockIdx.x;
  int lane = threadIdx.x & 31, wave = threadIdx.x >> 5;
  int m = lane & 15, hi = lane >> 4;
  int off = offsets[seg], cnt = counts[seg];
  int ntiles = (cnt + 15) >> 4;
  float vmax[4][8];
#pragma unroll
  for (int i = 0; i < 4; ++i)
#pragma unroll
    for (int r = 0; r < 8; ++r) vmax[i][r] = -__builtin_inff();

  for (int t = 0; t < ntiles; ++t) {
    int base = t * 16;
    int rm = base + m;
    if (rm >= cnt) rm = cnt - 1;                 // clamp tail rows
    int p = order[off + rm];
    const float* xr = feats + (size_t)p * FEATN;
    FragU fa;
#pragma unroll
    for (int e = 0; e < 16; ++e) fa.us[e] = 0;   // K padded with zeros
#pragma unroll
    for (int e = 0; e < 8; ++e) {                // elems 0-7: K = 8*hi + e
      int k = 8 * hi + e;
      if (k < FEATN) fa.us[e] = f2bf(xr[k]);
    }
#pragma unroll
    for (int nt = 0; nt < 4; ++nt) {
      int ncol = (wave * 4 + nt) * 16;
      v16bf fb = load_bfrag(Wc, 32, ncol, 0, lane);
      v8f acc;
#pragma unroll
      for (int r = 0; r < 8; ++r) acc[r] = 0.f;
      acc = wmma_bf16(fa.bf, fb, acc);
      float bb = bc[ncol + m];
#pragma unroll
      for (int r = 0; r < 8; ++r) {
        int rg = base + 8 * hi + r;              // D row M = 8*hi + r
        int rgs = rg < cnt ? rg : cnt - 1;       // clamped store address
        float v = acc[r] + bb;
        vmax[nt][r] = fmaxf(vmax[nt][r], v);     // duplicates == real last row
        hsorted[(size_t)(off + rgs) * H2 + ncol + m] = f2bf(v);
      }
    }
  }
#pragma unroll
  for (int nt = 0; nt < 4; ++nt) {
    float vr = vmax[nt][0];
#pragma unroll
    for (int r = 1; r < 8; ++r) vr = fmaxf(vr, vmax[nt][r]);
    vr = fmaxf(vr, __shfl_xor(vr, 16, 32));      // combine M halves
    if (lane < 16)
      gmax[(size_t)seg * H2 + (wave * 4 + nt) * 16 + m] = f2bf(vr);
  }
}

// ---------------- gpart = gmax @ W3a.T + b3 (2048x256 x 256x512) ------------
__global__ void __launch_bounds__(128) k_gpart(
    const unsigned short* __restrict__ gmax, const unsigned short* __restrict__ W3a,
    const float* __restrict__ b3, float* __restrict__ gpart) {
  int lane = threadIdx.x & 31, wave = threadIdx.x >> 5;
  int m = lane & 15, hi = lane >> 4;
  int tile = blockIdx.x * 4 + wave;              // 128 row tiles of 16 segments
  int rowbase = tile * 16;
  v16bf af[8];
#pragma unroll
  for (int kk = 0; kk < 8; ++kk)
    af[kk] = load_afrag(gmax, H2, (size_t)(rowbase + m), kk * 32, lane);
  for (int nt = 0; nt < 32; ++nt) {
    int ncol = nt * 16;
    v8f acc;
#pragma unroll
    for (int r = 0; r < 8; ++r) acc[r] = 0.f;
#pragma unroll
    for (int kk = 0; kk < 8; ++kk) {
      v16bf fb = load_bfrag(W3a, H2, ncol, kk * 32, lane);
      acc = wmma_bf16(af[kk], fb, acc);
    }
    float bb = b3[ncol + m];
#pragma unroll
    for (int r = 0; r < 8; ++r)
      gpart[(size_t)(rowbase + 8 * hi + r) * H3 + ncol + m] = acc[r] + bb;
  }
}

// ---------------- stage 2a: z = relu(gpart[seg] + h @ W3b.T) ---------------
// M=32 per wave: two A-fragment sets share each B fragment (halves W3b L2
// traffic); BN = 400000 is an exact multiple of 32 -> no tail masking.
__global__ void __launch_bounds__(128) k_stage2a(
    const unsigned short* __restrict__ hsorted, const int* __restrict__ segsorted,
    const float* __restrict__ gpart, const unsigned short* __restrict__ W3b,
    unsigned short* __restrict__ zsorted) {
  int lane = threadIdx.x & 31, wave = threadIdx.x >> 5;
  int m = lane & 15, hi = lane >> 4;
  int pair = blockIdx.x * 4 + wave;              // BN/32 = 12500 supertiles
  int rowbase = pair * 32;
  if (rowbase >= BN) return;
  v16bf af0[8], af1[8];
#pragma unroll
  for (int kk = 0; kk < 8; ++kk) {
    af0[kk] = load_afrag(hsorted, H2, (size_t)(rowbase + m), kk * 32, lane);
    af1[kk] = load_afrag(hsorted, H2, (size_t)(rowbase + 16 + m), kk * 32, lane);
  }
  int sr0[8], sr1[8];
#pragma unroll
  for (int r = 0; r < 8; ++r) {
    sr0[r] = segsorted[rowbase + 8 * hi + r];
    sr1[r] = segsorted[rowbase + 16 + 8 * hi + r];
  }
  for (int nt = 0; nt < 32; ++nt) {
    int ncol = nt * 16;
    v8f acc0, acc1;
#pragma unroll
    for (int r = 0; r < 8; ++r) { acc0[r] = 0.f; acc1[r] = 0.f; }
#pragma unroll
    for (int kk = 0; kk < 8; ++kk) {
      v16bf fb = load_bfrag(W3b, H2, ncol, kk * 32, lane);
      acc0 = wmma_bf16(af0[kk], fb, acc0);
      acc1 = wmma_bf16(af1[kk], fb, acc1);
    }
#pragma unroll
    for (int r = 0; r < 8; ++r) {
      float v0 = fmaxf(acc0[r] + gpart[(size_t)sr0[r] * H3 + ncol + m], 0.f);
      float v1 = fmaxf(acc1[r] + gpart[(size_t)sr1[r] * H3 + ncol + m], 0.f);
      zsorted[(size_t)(rowbase + 8 * hi + r) * H3 + ncol + m] = f2bf(v0);
      zsorted[(size_t)(rowbase + 16 + 8 * hi + r) * H3 + ncol + m] = f2bf(v1);
    }
  }
}

// ---------------- stage 2b: out = segmax(z @ W4.T) + b4 --------------------
// One WG (4 waves) per segment; waves partition the 384 outputs (6 n-tiles).
// 32-row A-supertile staged in LDS (async-to-LDS when available) and shared
// by all waves; each B fragment feeds two WMMAs. Tail rows clamped (see
// stage1 note) -> branch-free accumulation.
__global__ void __launch_bounds__(128) k_stage2b(
    const unsigned short* __restrict__ zsorted, const int* __restrict__ offsets,
    const int* __restrict__ counts, const unsigned short* __restrict__ W4b,
    const float* __restrict__ b4, float* __restrict__ out) {
  __shared__ unsigned short As[32 * H3];         // 32 KB
  int seg = blockIdx.x;
  int lane = threadIdx.x & 31, wave = threadIdx.x >> 5;
  int m = lane & 15, hi = lane >> 4;
  int off = offsets[seg], cnt = counts[seg];
  int nsuper = (cnt + 31) >> 5;
  float vmax[6][8];
#pragma unroll
  for (int i = 0; i < 6; ++i)
#pragma unroll
    for (int r = 0; r < 8; ++r) vmax[i][r] = -__builtin_inff();

  for (int t = 0; t < nsuper; ++t) {
    int base = t * 32;
    // cooperative 32x512 bf16 A-supertile -> LDS (2048 x 16B chunks)
#pragma unroll
    for (int j = 0; j < 16; ++j) {
      int c = threadIdx.x + 128 * j;             // 0..2047
      int rowl = c >> 6, cch = c & 63;
      int rg = base + rowl;
      rg = rg < cnt ? rg : cnt - 1;              // clamp tail rows
      const unsigned short* src = zsorted + (size_t)(off + rg) * H3 + cch * 8;
#ifdef USE_ASYNC_LDS
      __builtin_amdgcn_global_load_async_to_lds_b128(
          (as1_v4i_ptr)(void*)src, (as3_v4i_ptr)(void*)&As[c * 8], 0, 0);
#else
      *(v4u*)&As[c * 8] = *(const v4u*)src;
#endif
    }
#ifdef USE_ASYNC_LDS
    __builtin_amdgcn_s_wait_asynccnt(0);
#endif
    __syncthreads();

#pragma unroll
    for (int nt = 0; nt < 6; ++nt) {
      int ncol = (wave * 6 + nt) * 16;
      v8f acc0, acc1;
#pragma unroll
      for (int r = 0; r < 8; ++r) { acc0[r] = 0.f; acc1[r] = 0.f; }
#pragma unroll
      for (int kk = 0; kk < 16; ++kk) {
        v16bf fb  = load_bfrag(W4b, H3, ncol, kk * 32, lane);
        v16bf af0 = load_afrag(As, H3, (size_t)m, kk * 32, lane);
        v16bf af1 = load_afrag(As, H3, (size_t)(16 + m), kk * 32, lane);
        acc0 = wmma_bf16(af0, fb, acc0);
        acc1 = wmma_bf16(af1, fb, acc1);
      }
#pragma unroll
      for (int r = 0; r < 8; ++r) {              // duplicates == real last row
        vmax[nt][r] = fmaxf(vmax[nt][r], fmaxf(acc0[r], acc1[r]));
      }
    }
    __syncthreads();                             // before LDS overwrite
  }
#pragma unroll
  for (int nt = 0; nt < 6; ++nt) {
    float vr = vmax[nt][0];
#pragma unroll
    for (int r = 1; r < 8; ++r) vr = fmaxf(vr, vmax[nt][r]);
    vr = fmaxf(vr, __shfl_xor(vr, 16, 32));
    if (lane < 16) {
      int col = (wave * 6 + nt) * 16 + m;
      out[(size_t)seg * EOUT + col] = vr + b4[col];  // max(y)+b4 == max(y+b4)
    }
  }
}

// ---------------- host launcher ----------------
extern "C" void kernel_launch(void* const* d_in, const int* in_sizes, int n_in,
                              void* d_out, int out_size, void* d_ws, size_t ws_size,
                              hipStream_t stream) {
  const float* feats = (const float*)d_in[0];
  const int*   sidx  = (const int*)d_in[1];
  const float* W1 = (const float*)d_in[2];
  const float* b1 = (const float*)d_in[3];
  const float* W2 = (const float*)d_in[4];
  const float* b2 = (const float*)d_in[5];
  const float* W3 = (const float*)d_in[6];
  const float* b3 = (const float*)d_in[7];
  const float* W4 = (const float*)d_in[8];
  const float* b4 = (const float*)d_in[9];
  float* out = (float*)d_out;

  uint8_t* wsb = (uint8_t*)d_ws;
  size_t o = 0;
  auto alloc = [&](size_t bytes) -> void* {
    o = (o + 255) & ~(size_t)255;
    void* p = wsb + o;
    o += bytes;
    return p;
  };
  int* counts    = (int*)alloc(SEGS * 4);
  int* cursor    = (int*)alloc(SEGS * 4);
  int* offsets   = (int*)alloc((SEGS + 1) * 4);
  int* order     = (int*)alloc((size_t)BN * 4);
  int* segsorted = (int*)alloc((size_t)BN * 4);
  unsigned short* Wc    = (unsigned short*)alloc(256 * 32 * 2);
  float*          bc    = (float*)alloc(256 * 4);
  unsigned short* W3a   = (unsigned short*)alloc((size_t)H3 * H2 * 2);
  unsigned short* W3b   = (unsigned short*)alloc((size_t)H3 * H2 * 2);
  unsigned short* W4b   = (unsigned short*)alloc((size_t)EOUT * H3 * 2);
  unsigned short* gmaxb = (unsigned short*)alloc((size_t)SEGS * H2 * 2);
  float*          gpart = (float*)alloc((size_t)SEGS * H3 * 4);
  unsigned short* hsorted = (unsigned short*)alloc((size_t)BN * H2 * 2);
  unsigned short* zsorted = (unsigned short*)alloc((size_t)BN * H3 * 2);

  k_zero<<<(SEGS + 255) / 256, 256, 0, stream>>>(counts, SEGS);
  k_zero<<<(SEGS + 255) / 256, 256, 0, stream>>>(cursor, SEGS);
  k_count<<<(BN + 255) / 256, 256, 0, stream>>>(sidx, counts);
  k_scan<<<1, 256, 0, stream>>>(counts, offsets);
  k_scatter<<<(BN + 255) / 256, 256, 0, stream>>>(sidx, offsets, cursor, order, segsorted);
  k_fuse12<<<(256 * 32 + 255) / 256, 256, 0, stream>>>(W1, b1, W2, b2, Wc, bc);
  k_cvt<<<(H3 * H2 + 255) / 256, 256, 0, stream>>>(W3, H3, H2, H3 * H2, W3a);
  k_cvt<<<(H3 * H2 + 255) / 256, 256, 0, stream>>>(W3 + 256, H3, H2, H3 * H2, W3b);
  k_cvt<<<(EOUT * H3 + 255) / 256, 256, 0, stream>>>(W4, H3, H3, EOUT * H3, W4b);

  k_stage1<<<SEGS, 128, 0, stream>>>(feats, order, offsets, counts, Wc, bc, hsorted, gmaxb);
  k_gpart<<<32, 128, 0, stream>>>(gmaxb, W3a, b3, gpart);
  k_stage2a<<<BN / 32 / 4, 128, 0, stream>>>(hsorted, segsorted, gpart, W3b, zsorted);
  k_stage2b<<<SEGS, 128, 0, stream>>>(zsorted, offsets, counts, W4b, b4, out);
}
